// ATK_SPA_87351044866230
// MI455X (gfx1250) — matrix operand order repaired and checked by reading, not compiled, not run on hardware
//
#include <hip/hip_runtime.h>
#include <math.h>

typedef _Float16 htype;
typedef __attribute__((ext_vector_type(16))) _Float16 v16h;
typedef __attribute__((ext_vector_type(8)))  _Float16 v8h;
typedef __attribute__((ext_vector_type(8)))  float    v8f;
typedef __attribute__((ext_vector_type(4)))  unsigned int u32x4;
typedef __attribute__((ext_vector_type(4)))  int i32x4;
typedef __attribute__((ext_vector_type(8)))  int i32x8;

#define NBATCH 2
#define HH     256
#define WW     256
#define DIMC   256
#define PD     128
#define NHEADS 8
#define CHD    16
#define NPB    (HH*WW)        /* 65536 pixels per batch image */
#define NPIX   (NBATCH*NPB)   /* 131072 total pixels */

// ---- Tensor Data Mover plumbing (arity differs per toolchain; probe-informed)
#if __has_builtin(__builtin_amdgcn_tensor_load_to_lds)
  #define HAVE_TDM 1
  #if __has_include(<hip/amd_detail/amd_gfx1250_TDM.h>)
    // amdgpu-toolchain (clang-23): 6-arg form
    #define TDM_LOAD(g0, g1, g2, g3) \
      __builtin_amdgcn_tensor_load_to_lds((g0), (g1), (g2), (g3), (i32x8){0,0,0,0,0,0,0,0}, 0)
  #else
    // ROCm 7.2 (clang-22): 5-arg form
    #define TDM_LOAD(g0, g1, g2, g3) \
      __builtin_amdgcn_tensor_load_to_lds((g0), (g1), (g2), (g3), 0)
  #endif
#endif

#if __has_builtin(__builtin_amdgcn_s_wait_tensorcnt)
  #define WAIT_TENSOR() __builtin_amdgcn_s_wait_tensorcnt(0)
#else
  #define WAIT_TENSOR() asm volatile("s_wait_tensorcnt 0x0" ::: "memory")
#endif

static __device__ __forceinline__ v8f wmma_f16(v16h a, v16h b, v8f c) {
  // D = A(16x32 f16) * B(32x16 f16) + C(16x16 f32)
  return __builtin_amdgcn_wmma_f32_16x16x32_f16(false, a, false, b, (short)0, c,
                                                false, false);
}

// Issue a TDM load of a [16 x tileK] f16 tile (row stride 256 halves) into LDS.
#ifdef HAVE_TDM
static __device__ __forceinline__ void tdm_load_tile(const htype* gsrc, void* lds,
                                                     unsigned tileK) {
  unsigned long long ga = (unsigned long long)(const void*)gsrc;
  unsigned ldsOff = (unsigned)(unsigned long long)lds;   // low 32 bits = LDS byte offset
  u32x4 g0;
  g0[0] = 1u;                                            // count=1, user descriptor
  g0[1] = ldsOff;                                        // lds_addr (bytes)
  g0[2] = (unsigned)ga;                                  // global_addr[31:0]
  g0[3] = (unsigned)((ga >> 32) & 0x01FFFFFFu) | (2u << 30);  // addr[56:32] | type=2
  i32x8 g1;
  g1[0] = (int)(1u << 16);                 // data_size=1 (2 bytes), wg_mask=0, no pad
  g1[1] = (int)(256u << 16);               // tensor_dim0 = 256 (low 16 bits)
  g1[2] = 0;                               // tensor_dim0 hi | tensor_dim1 lo
  g1[3] = (int)(2u | (tileK << 16));       // tensor_dim1 hi (131072>>16=2) | tile_dim0
  g1[4] = 16;                              // tile_dim1 = 16 rows, tile_dim2 = 0
  g1[5] = 256;                             // tensor_dim0_stride = 256 (lo 32)
  g1[6] = 0;                               // stride0 hi | tensor_dim1_stride lo
  g1[7] = 0;
  i32x4 gz = {0, 0, 0, 0};
  TDM_LOAD(g0, g1, gz, gz);
  WAIT_TENSOR();
}
#endif

// ---------------------------------------------------------------- converts
__global__ void k_cvt_x(const float* __restrict__ x, htype* __restrict__ xh) {
  size_t i = (size_t)blockIdx.x * 256 + threadIdx.x;   // grid covers NPIX*256 exactly
  xh[i] = (htype)x[i];
}

__global__ void k_cvt_w(const float* __restrict__ qw, const float* __restrict__ pw,
                        const float* __restrict__ gw,
                        htype* __restrict__ wq, htype* __restrict__ wp,
                        htype* __restrict__ wg) {
  int i = blockIdx.x * 256 + threadIdx.x;              // grid covers 65536 exactly
  if (i < 49152) wq[i] = (htype)qw[i];                 // qkv_w: 384x128
  wp[i] = (htype)pw[i];                                // proj_w: 256x256
  if (i < 32768) wg[i] = (htype)gw[i];                 // g1_w: 128x256
}

__global__ void k_zero(float* __restrict__ p, int n) {
  int i = blockIdx.x * 256 + threadIdx.x;
  if (i < n) p[i] = 0.f;
}

// -------------------------------------------------- qkv GEMM: [N,384]=X[N,128(of 256)]*Wq^T
// 6 waves/block; A-tile (16 px x 128 K, f16) staged once into LDS via TDM and
// shared across all 24 output-channel tiles (4 per wave).
__global__ void __launch_bounds__(192) k_qkv_gemm(const htype* __restrict__ X,
                                                  const htype* __restrict__ W,
                                                  htype* __restrict__ Y) {
  __shared__ htype sA[16 * 128];
  const int tid  = threadIdx.x;
  const int lane = tid & 31;
  const int wave = tid >> 5;          // 0..5
  const int col  = lane & 15;
  const bool hi  = lane >= 16;
  const size_t nb = (size_t)blockIdx.x * 16;
#ifdef HAVE_TDM
  if (wave == 0) tdm_load_tile(X + nb * 256, sA, 128u);
  __syncthreads();
#else
  for (int i = tid; i < 16 * 128 / 8; i += 192)
    *(v8h*)(sA + i * 8) = *(const v8h*)(X + nb * 256 + (i / 16) * 256 + (i % 16) * 8);
  __syncthreads();
#endif
  const int ob0 = wave * 64;                       // 4 tiles of 16 per wave
  const htype* w0   = W + (size_t)(ob0 + col) * 128;
  const htype* arow = sA + col * 128;
  v8f acc0 = {}, acc1 = {}, acc2 = {}, acc3 = {};
#pragma unroll
  for (int k0 = 0; k0 < 128; k0 += 32) {
    const int ka = k0 + (hi ? 16 : 0);
    v8h a0 = *(const v8h*)(arow + ka);
    v8h a1 = *(const v8h*)(arow + ka + 8);
    v16h a;
#pragma unroll
    for (int i = 0; i < 8; ++i) { a[i] = a0[i]; a[i + 8] = a1[i]; }
    const int kb = k0 + (hi ? 16 : 0);
    v16h b0 = *(const v16h*)(w0 + kb);
    v16h b1 = *(const v16h*)(w0 + 16 * 128 + kb);
    v16h b2 = *(const v16h*)(w0 + 32 * 128 + kb);
    v16h b3 = *(const v16h*)(w0 + 48 * 128 + kb);
    __builtin_prefetch(w0 + kb + 32, 0, 1);
    acc0 = wmma_f16(a, b0, acc0);
    acc1 = wmma_f16(a, b1, acc1);
    acc2 = wmma_f16(a, b2, acc2);
    acc3 = wmma_f16(a, b3, acc3);
  }
#pragma unroll
  for (int r = 0; r < 8; ++r) {
    const int M = r + (hi ? 8 : 0);
    htype* yr = Y + (nb + M) * 384 + col;
    yr[ob0]      = (htype)acc0[r];
    yr[ob0 + 16] = (htype)acc1[r];
    yr[ob0 + 32] = (htype)acc2[r];
    yr[ob0 + 48] = (htype)acc3[r];
  }
}

// -------------------------------------------------- gating: g1 GEMM + relu + g2 dot + sigmoid + mean
__global__ void k_gate(const htype* __restrict__ X, const htype* __restrict__ Wg,
                       const float* __restrict__ g1b, const float* __restrict__ g2w,
                       const float* __restrict__ g2bp, float* __restrict__ gsum) {
  const int lane = threadIdx.x;
  const int col  = lane & 15;
  const bool hi  = lane >= 16;
  const size_t nb = (size_t)blockIdx.x * 16;
  const htype* xrow = X + (nb + col) * 256;
  v8f acc[8] = {};
  for (int k0 = 0; k0 < 256; k0 += 32) {
    const int ka = k0 + (hi ? 16 : 0);
    v8h a0 = *(const v8h*)(xrow + ka);
    v8h a1 = *(const v8h*)(xrow + ka + 8);
    v16h a;
#pragma unroll
    for (int i = 0; i < 8; ++i) { a[i] = a0[i]; a[i + 8] = a1[i]; }
    const int kb = k0 + (hi ? 16 : 0);
#pragma unroll
    for (int t = 0; t < 8; ++t) {
      v16h b = *(const v16h*)(Wg + (size_t)(t * 16 + col) * 256 + kb);
      acc[t] = wmma_f16(a, b, acc[t]);
    }
  }
  float g1bv[8], g2wv[8];
#pragma unroll
  for (int t = 0; t < 8; ++t) {
    g1bv[t] = g1b[t * 16 + col];
    g2wv[t] = g2w[t * 16 + col];
  }
  const float g2b0 = g2bp[0];
  float gacc = 0.f;
#pragma unroll
  for (int r = 0; r < 8; ++r) {
    float p = 0.f;
#pragma unroll
    for (int t = 0; t < 8; ++t) {
      float v = acc[t][r] + g1bv[t];
      v = fmaxf(v, 0.f);
      p += v * g2wv[t];
    }
    for (int off = 8; off; off >>= 1) p += __shfl_xor(p, off, 16);
    if (col == 0) gacc += 1.f / (1.f + expf(-(p + g2b0)));
  }
  if (col == 0) atomicAdd(gsum, gacc);
}

// -------------------------------------------------- depthwise 3x3, channels-last, 8 ch / thread
__global__ void k_dwconv(const htype* __restrict__ qkvh, const float* __restrict__ dww,
                         htype* __restrict__ qkvd) {
  const int gid = blockIdx.x * 256 + threadIdx.x;       // NPIX*48 threads
  const int grp = gid % 48;
  const int n   = gid / 48;
  const int bb  = n / NPB;
  const int rem = n % NPB;
  const int y   = rem / WW;
  const int x   = rem % WW;
  const int cb  = grp * 8;
  float accv[8] = {0.f, 0.f, 0.f, 0.f, 0.f, 0.f, 0.f, 0.f};
#pragma unroll
  for (int dy = -1; dy <= 1; ++dy) {
    const int yy = y + dy;
    if (yy < 0 || yy >= HH) continue;
#pragma unroll
    for (int dx = -1; dx <= 1; ++dx) {
      const int xx = x + dx;
      if (xx < 0 || xx >= WW) continue;
      const v8h d = *(const v8h*)(qkvh + ((size_t)(bb * NPB + yy * WW + xx)) * 384 + cb);
      const int tap = (dy + 1) * 3 + (dx + 1);
#pragma unroll
      for (int i = 0; i < 8; ++i) accv[i] += (float)d[i] * dww[(cb + i) * 9 + tap];
    }
  }
  v8h o;
#pragma unroll
  for (int i = 0; i < 8; ++i) o[i] = (htype)accv[i];
  *(v8h*)(qkvd + (size_t)n * 384 + cb) = o;
}

// -------------------------------------------------- per-(b,chan) sum-of-squares over 65536 pixels
__global__ void k_norm(const htype* __restrict__ qkvd, float* __restrict__ norms) {
  __shared__ float red[256];
  const int bj = blockIdx.x;        // 0..511 : b*256 + chan (q: 0..127, k: 128..255)
  const int b  = bj >> 8;
  const int j  = bj & 255;
  const htype* base = qkvd + (size_t)b * NPB * 384 + j;
  float s = 0.f;
  for (int n = threadIdx.x; n < NPB; n += 256) {
    const float v = (float)base[(size_t)n * 384];
    s += v * v;
  }
  red[threadIdx.x] = s;
  __syncthreads();
  for (int st = 128; st; st >>= 1) {
    if (threadIdx.x < st) red[threadIdx.x] += red[threadIdx.x + st];
    __syncthreads();
  }
  if (threadIdx.x == 0) norms[bj] = red[0];
}

// -------------------------------------------------- Gram S[c,d] = sum_n q[c,n]*k[d,n] via WMMA
__global__ void k_gram(const htype* __restrict__ qkvd, float* __restrict__ attnS) {
  const int lane = threadIdx.x;
  const int col  = lane & 15;
  const bool hi  = lane >= 16;
  const int bh   = blockIdx.x;           // 0..15
  const int b    = bh >> 3;
  const int head = bh & 7;
  const size_t n0 = (size_t)blockIdx.y * 512;   // 128 chunks over 65536
  const htype* qb = qkvd + (size_t)b * NPB * 384 + head * 16 + col;        // q[c=col, n]
  const htype* kp = qb + 128;                                               // k[d=col, n]
  const int ka = hi ? 16 : 0;
  const int kk = hi ? 16 : 0;
  v8f acc = {};
  for (int it = 0; it < 16; ++it) {
    const size_t n = n0 + (size_t)it * 32;
    v16h a, bm;
#pragma unroll
    for (int i = 0; i < 8; ++i) {
      a[i]     = qb[(n + ka + i) * 384];
      a[i + 8] = qb[(n + ka + 8 + i) * 384];
    }
#pragma unroll
    for (int i = 0; i < 16; ++i) bm[i] = kp[(n + kk + i) * 384];
    acc = wmma_f16(a, bm, acc);
  }
  float* dst = attnS + bh * 256;
#pragma unroll
  for (int r = 0; r < 8; ++r) {
    const int M = r + (hi ? 8 : 0);
    atomicAdd(dst + M * 16 + col, acc[r]);
  }
}

// -------------------------------------------------- normalize, temperature, top-k mask, softmax
__global__ void k_attn_final(const float* __restrict__ attnS, const float* __restrict__ norms,
                             const float* __restrict__ gsum, const float* __restrict__ temp,
                             float* __restrict__ attnF) {
  const int t = threadIdx.x;            // 256 threads = 2*8*16 rows
  const int bh = t >> 4;
  const int c  = t & 15;
  const int b  = bh >> 3;
  const int head = bh & 7;
  const float dynk = floorf(16.f * (gsum[0] / (float)NPIX));
  const float qn = fmaxf(sqrtf(norms[b * 256 + head * 16 + c]), 1e-12f);
  const float tv = temp[head];
  float row[16];
#pragma unroll
  for (int d = 0; d < 16; ++d) {
    const float kn = fmaxf(sqrtf(norms[b * 256 + 128 + head * 16 + d]), 1e-12f);
    row[d] = attnS[bh * 256 + c * 16 + d] / (qn * kn) * tv;
  }
  float masked[16];
#pragma unroll
  for (int d = 0; d < 16; ++d) {
    int rk = 0;
#pragma unroll
    for (int e = 0; e < 16; ++e)
      rk += (row[e] > row[d]) || (row[e] == row[d] && e < d);
    masked[d] = ((float)rk < dynk) ? row[d] : -INFINITY;
  }
  float mx = -INFINITY;
#pragma unroll
  for (int d = 0; d < 16; ++d) mx = fmaxf(mx, masked[d]);
  float se = 0.f, ex[16];
#pragma unroll
  for (int d = 0; d < 16; ++d) { ex[d] = expf(masked[d] - mx); se += ex[d]; }
#pragma unroll
  for (int d = 0; d < 16; ++d) attnF[bh * 256 + c * 16 + d] = ex[d] / se;
}

// -------------------------------------------------- z = [0.8 * attn@v  |  x2] in f16
__global__ void k_zbuild(const htype* __restrict__ qkvd, const float* __restrict__ attnF,
                         const float* __restrict__ x,
                         const float* __restrict__ a1, const float* __restrict__ a2,
                         const float* __restrict__ a3, const float* __restrict__ a4,
                         htype* __restrict__ zh) {
  const int gid = blockIdx.x * 256 + threadIdx.x;   // NPIX*8 threads
  const int head = gid & 7;
  const size_t n = (size_t)(gid >> 3);
  const int b = (int)(n / NPB);
  const htype* vrow = qkvd + n * 384 + 256 + head * 16;
  float vv[16];
#pragma unroll
  for (int d = 0; d < 16; ++d) vv[d] = (float)vrow[d];
  const float* A = attnF + (size_t)(b * 8 + head) * 256;
  const float sc = a1[0] + a2[0] + a3[0] + a4[0];
  htype* zr = zh + n * 256;
#pragma unroll
  for (int c = 0; c < 16; ++c) {
    float s = 0.f;
#pragma unroll
    for (int d = 0; d < 16; ++d) s += A[c * 16 + d] * vv[d];
    zr[head * 16 + c] = (htype)(s * sc);
  }
  const float* xr = x + n * 256 + 128 + head * 16;
#pragma unroll
  for (int i = 0; i < 16; ++i) zr[128 + head * 16 + i] = (htype)xr[i];
}

// -------------------------------------------------- proj GEMM: out[N,256] = Z[N,256]*Wp^T (f32 out)
// 4 waves/block; A-tile (16 px x 256 K, f16) staged once via TDM into LDS,
// shared across all 16 output tiles (4 per wave).
__global__ void __launch_bounds__(128) k_proj_gemm(const htype* __restrict__ X,
                                                   const htype* __restrict__ W,
                                                   float* __restrict__ Y) {
  __shared__ htype sA[16 * 256];
  const int tid  = threadIdx.x;
  const int lane = tid & 31;
  const int wave = tid >> 5;          // 0..3
  const int col  = lane & 15;
  const bool hi  = lane >= 16;
  const size_t nb = (size_t)blockIdx.x * 16;
#ifdef HAVE_TDM
  if (wave == 0) tdm_load_tile(X + nb * 256, sA, 256u);
  __syncthreads();
#else
  for (int i = tid; i < 16 * 256 / 8; i += 128)
    *(v8h*)(sA + i * 8) = *(const v8h*)(X + nb * 256 + i * 8);
  __syncthreads();
#endif
  const int ob0 = wave * 64;                       // 4 tiles of 16 per wave
  const htype* w0   = W + (size_t)(ob0 + col) * 256;
  const htype* arow = sA + col * 256;
  v8f acc0 = {}, acc1 = {}, acc2 = {}, acc3 = {};
#pragma unroll
  for (int k0 = 0; k0 < 256; k0 += 32) {
    const int ka = k0 + (hi ? 16 : 0);
    v8h a0 = *(const v8h*)(arow + ka);
    v8h a1 = *(const v8h*)(arow + ka + 8);
    v16h a;
#pragma unroll
    for (int i = 0; i < 8; ++i) { a[i] = a0[i]; a[i + 8] = a1[i]; }
    const int kb = k0 + (hi ? 16 : 0);
    v16h b0 = *(const v16h*)(w0 + kb);
    v16h b1 = *(const v16h*)(w0 + 16 * 256 + kb);
    v16h b2 = *(const v16h*)(w0 + 32 * 256 + kb);
    v16h b3 = *(const v16h*)(w0 + 48 * 256 + kb);
    __builtin_prefetch(w0 + kb + 32, 0, 1);
    acc0 = wmma_f16(a, b0, acc0);
    acc1 = wmma_f16(a, b1, acc1);
    acc2 = wmma_f16(a, b2, acc2);
    acc3 = wmma_f16(a, b3, acc3);
  }
#pragma unroll
  for (int r = 0; r < 8; ++r) {
    const int M = r + (hi ? 8 : 0);
    float* yr = Y + (nb + M) * 256 + col;
    yr[ob0]      = acc0[r];
    yr[ob0 + 16] = acc1[r];
    yr[ob0 + 32] = acc2[r];
    yr[ob0 + 48] = acc3[r];
  }
}

// ================================================================ launcher
extern "C" void kernel_launch(void* const* d_in, const int* in_sizes, int n_in,
                              void* d_out, int out_size, void* d_ws, size_t ws_size,
                              hipStream_t stream) {
  (void)in_sizes; (void)n_in; (void)out_size; (void)ws_size;
  const float* x     = (const float*)d_in[0];
  const float* qkvw  = (const float*)d_in[1];
  const float* dww   = (const float*)d_in[2];
  const float* projw = (const float*)d_in[3];
  const float* g1w   = (const float*)d_in[4];
  const float* g1b   = (const float*)d_in[5];
  const float* g2w   = (const float*)d_in[6];
  const float* g2b   = (const float*)d_in[7];
  const float* temp  = (const float*)d_in[8];
  const float* a1    = (const float*)d_in[9];
  const float* a2    = (const float*)d_in[10];
  const float* a3    = (const float*)d_in[11];
  const float* a4    = (const float*)d_in[12];
  float* out = (float*)d_out;

  char* ws = (char*)d_ws;
  size_t off = 0;
  htype* xch  = (htype*)(ws + off); off += (size_t)NPIX * 256 * sizeof(htype);
  htype* qkvh = (htype*)(ws + off); off += (size_t)NPIX * 384 * sizeof(htype);
  htype* qkvd = (htype*)(ws + off); off += (size_t)NPIX * 384 * sizeof(htype);
  htype* zh   = (htype*)(ws + off); off += (size_t)NPIX * 256 * sizeof(htype);
  htype* wq   = (htype*)(ws + off); off += (size_t)49152 * sizeof(htype);
  htype* wp   = (htype*)(ws + off); off += (size_t)65536 * sizeof(htype);
  htype* wg   = (htype*)(ws + off); off += (size_t)32768 * sizeof(htype);
  off = (off + 255) & ~(size_t)255;
  float* norms = (float*)(ws + off);            // 512
  float* attnS = norms + 512;                   // 4096
  float* attnF = attnS + 4096;                  // 4096
  float* gsum  = attnF + 4096;                  // 1
  const int accFloats = 512 + 4096 + 4096 + 1;

  k_cvt_x<<<NPIX * 256 / 256, 256, 0, stream>>>(x, xch);
  k_cvt_w<<<65536 / 256, 256, 0, stream>>>(qkvw, projw, g1w, wq, wp, wg);
  k_zero<<<(accFloats + 255) / 256, 256, 0, stream>>>(norms, accFloats);

  k_qkv_gemm<<<NPIX / 16, 192, 0, stream>>>(xch, wq, qkvh);
  k_gate<<<NPIX / 16, 32, 0, stream>>>(xch, wg, g1b, g2w, g2b, gsum);
  k_dwconv<<<NPIX * 48 / 256, 256, 0, stream>>>(qkvh, dww, qkvd);
  k_norm<<<512, 256, 0, stream>>>(qkvd, norms);
  k_gram<<<dim3(16, 128), 32, 0, stream>>>(qkvd, attnS);
  k_attn_final<<<1, 256, 0, stream>>>(attnS, norms, gsum, temp, attnF);
  k_zbuild<<<NPIX * 8 / 256, 256, 0, stream>>>(qkvd, attnF, x, a1, a2, a3, a4, zh);
  k_proj_gemm<<<dim3(NPIX / 16), 128, 0, stream>>>(zh, wp, out);
}
